// GDANet_cls_40913858462410
// MI455X (gfx1250) — matrix operand order repaired and checked by reading, not compiled, not run on hardware
//
#include <hip/hip_runtime.h>
#include <hip/hip_bf16.h>
#include <float.h>

typedef __attribute__((ext_vector_type(16))) _Float16 v16h;
typedef __attribute__((ext_vector_type(8)))  float    v8f;

#define NPTS 1024
#define NB   16
#define MGDM 256

// ---------------- WMMA fragment loaders (f32 -> f16 convert on the fly) ---------------
// A (16x32 f16): lane<16 holds row M=lane, VGPR i in [0,3]: K=2i(+8 if lane>=16),
//                i in [4,7]: K=16+2(i-4)(+8 if lane>=16)
__device__ __forceinline__ int a_kbase(int i, int half) {
  return (i < 4) ? (2 * i + half * 8) : (16 + 2 * (i - 4) + half * 8);
}

// Unconditional (fast-path) loaders: Arow/Bcol are pre-offset to the (clamped) row/col.
__device__ __forceinline__ v16h load_a_fast(const float* __restrict__ Arow, int sak,
                                            int k0, int half) {
  v16h f;
#pragma unroll
  for (int i = 0; i < 8; ++i) {
    int k = k0 + a_kbase(i, half);
    f[2 * i]     = (_Float16)Arow[k * sak];
    f[2 * i + 1] = (_Float16)Arow[(k + 1) * sak];
  }
  return f;
}

__device__ __forceinline__ v16h load_b_fast(const float* __restrict__ Bcol, int sbk,
                                            int k0, int half) {
  v16h f;
#pragma unroll
  for (int i = 0; i < 8; ++i) {
    int k = k0 + a_kbase(i, half);
    f[2 * i]     = (_Float16)Bcol[k * sbk];
    f[2 * i + 1] = (_Float16)Bcol[(k + 1) * sbk];
  }
  return f;
}

// Tail loaders: k-range may exceed K. Clamp the index (always-legal load) and zero the
// value with a select -- no exec-mask branching.
__device__ __forceinline__ v16h load_a_tail(const float* __restrict__ Arow, int sak,
                                            int k0, int K, int half) {
  v16h f;
#pragma unroll
  for (int i = 0; i < 8; ++i) {
    int k = k0 + a_kbase(i, half);
    int kc0 = (k < K) ? k : 0;
    int kc1 = (k + 1 < K) ? (k + 1) : 0;
    float a0 = Arow[kc0 * sak];
    float a1 = Arow[kc1 * sak];
    f[2 * i]     = (_Float16)((k < K) ? a0 : 0.0f);
    f[2 * i + 1] = (_Float16)((k + 1 < K) ? a1 : 0.0f);
  }
  return f;
}

__device__ __forceinline__ v16h load_b_tail(const float* __restrict__ Bcol, int sbk,
                                            int k0, int K, int half) {
  v16h f;
#pragma unroll
  for (int i = 0; i < 8; ++i) {
    int k = k0 + a_kbase(i, half);
    int kc0 = (k < K) ? k : 0;
    int kc1 = (k + 1 < K) ? (k + 1) : 0;
    float b0 = Bcol[kc0 * sbk];
    float b1 = Bcol[kc1 * sbk];
    f[2 * i]     = (_Float16)((k < K) ? b0 : 0.0f);
    f[2 * i + 1] = (_Float16)((k + 1 < K) ? b1 : 0.0f);
  }
  return f;
}

// ---------------- Generic strided WMMA GEMM with epilogue -----------------------------
// C[m,n] = relu?( bns[m]*(alpha*sum_k A[m,k]B[k,n] + bias[m]) + bnb[m] ) + R[m,n]
// Each wave computes a 16x32 tile (two 16x16 WMMA tiles sharing one A fragment).
// Intra-matrix strides are 32-bit; only batch strides are 64-bit.
__global__ void gda_wmma_gemm_kernel(
    const float* __restrict__ A, int sam, int sak, long sab,
    const float* __restrict__ Bp, int sbk, int sbn, long sbb,
    float* __restrict__ C, int scm, int scn, long scb,
    const float* __restrict__ bias,
    const float* __restrict__ bns, const float* __restrict__ bnb,
    const float* __restrict__ R, int srm, int srn, long srb,
    int M, int N, int K, float alpha, int relu)
{
  int tilesN  = (N + 15) >> 4;
  int tilesM  = (M + 15) >> 4;
  int tilesN2 = (tilesN + 1) >> 1;           // pairs of N-tiles
  int tile = blockIdx.x * blockDim.y + threadIdx.y;
  if (tile >= tilesM * tilesN2) return;
  int tm = tile / tilesN2, tn2 = tile % tilesN2;
  int lane = threadIdx.x;
  int half = lane >> 4, l15 = lane & 15;
  int m  = tm * 16 + l15;
  int n0 = tn2 * 32 + l15;
  int n1 = n0 + 16;
  // Clamp row/col: out-of-range lanes compute garbage rows that are never stored.
  int mc  = (m  < M) ? m  : (M - 1);
  int nc0 = (n0 < N) ? n0 : (N - 1);
  int nc1 = (n1 < N) ? n1 : (N - 1);
  const float* Arow  = A  + (long)blockIdx.y * sab + (long)mc  * sam;
  const float* Bcol0 = Bp + (long)blockIdx.y * sbb + (long)nc0 * sbn;
  const float* Bcol1 = Bp + (long)blockIdx.y * sbb + (long)nc1 * sbn;
  v8f acc0 = {0.f, 0.f, 0.f, 0.f, 0.f, 0.f, 0.f, 0.f};
  v8f acc1 = {0.f, 0.f, 0.f, 0.f, 0.f, 0.f, 0.f, 0.f};
  int Kmain = K & ~31;
  for (int k0 = 0; k0 < Kmain; k0 += 32) {
    __builtin_prefetch(Bcol0 + (k0 + 32) * sbk, 0, 3);
    v16h af  = load_a_fast(Arow,  sak, k0, half);
    v16h bf0 = load_b_fast(Bcol0, sbk, k0, half);
    v16h bf1 = load_b_fast(Bcol1, sbk, k0, half);
    acc0 = __builtin_amdgcn_wmma_f32_16x16x32_f16(false, af, false, bf0, (short)0, acc0,
                                                  false, false);
    acc1 = __builtin_amdgcn_wmma_f32_16x16x32_f16(false, af, false, bf1, (short)0, acc1,
                                                  false, false);
  }
  if (Kmain < K) {
    v16h af  = load_a_tail(Arow,  sak, Kmain, K, half);
    v16h bf0 = load_b_tail(Bcol0, sbk, Kmain, K, half);
    v16h bf1 = load_b_tail(Bcol1, sbk, Kmain, K, half);
    acc0 = __builtin_amdgcn_wmma_f32_16x16x32_f16(false, af, false, bf0, (short)0, acc0,
                                                  false, false);
    acc1 = __builtin_amdgcn_wmma_f32_16x16x32_f16(false, af, false, bf1, (short)0, acc1,
                                                  false, false);
  }
#pragma unroll
  for (int t = 0; t < 2; ++t) {
    int n = (t == 0) ? n0 : n1;
    v8f acc = (t == 0) ? acc0 : acc1;
    if (n < N) {
#pragma unroll
      for (int r = 0; r < 8; ++r) {
        int mo = tm * 16 + r + half * 8;
        if (mo < M) {
          float v = alpha * acc[r];
          if (bias) v += bias[mo];
          if (bns)  v = bns[mo] * v + bnb[mo];
          if (relu) v = fmaxf(v, 0.0f);
          if (R)    v += R[(long)blockIdx.y * srb + (long)mo * srm + (long)n * srn];
          C[(long)blockIdx.y * scb + (long)mo * scm + (long)n * scn] = v;
        }
      }
    }
  }
}

// p[m,n] = 2 * <x_m, x_n> - ||x_m||^2 - ||x_n||^2   (single batch; X is C x N row-major)
// M = N = NPTS (multiples of 32), K = C may need a tail chunk. 16x32 tile per wave.
__global__ void gda_pmat_kernel(const float* __restrict__ X, const float* __restrict__ xx,
                                float* __restrict__ P, int C, int N)
{
  int tiles  = N >> 4;
  int tiles2 = tiles >> 1;
  int tile = blockIdx.x * blockDim.y + threadIdx.y;
  if (tile >= tiles * tiles2) return;
  int tm = tile / tiles2, tn2 = tile % tiles2;
  int lane = threadIdx.x, half = lane >> 4, l15 = lane & 15;
  int m = tm * 16 + l15;
  int n0 = tn2 * 32 + l15, n1 = n0 + 16;
  const float* Arow  = X + m;    // A[m,k] = X[k*N+m] : row base X+m, stride N
  const float* Bcol0 = X + n0;   // B[k,n] = X[k*N+n] : col base X+n, stride N
  const float* Bcol1 = X + n1;
  v8f acc0 = {0.f, 0.f, 0.f, 0.f, 0.f, 0.f, 0.f, 0.f};
  v8f acc1 = {0.f, 0.f, 0.f, 0.f, 0.f, 0.f, 0.f, 0.f};
  int Kmain = C & ~31;
  for (int k0 = 0; k0 < Kmain; k0 += 32) {
    v16h af  = load_a_fast(Arow,  N, k0, half);
    v16h bf0 = load_b_fast(Bcol0, N, k0, half);
    v16h bf1 = load_b_fast(Bcol1, N, k0, half);
    acc0 = __builtin_amdgcn_wmma_f32_16x16x32_f16(false, af, false, bf0, (short)0, acc0,
                                                  false, false);
    acc1 = __builtin_amdgcn_wmma_f32_16x16x32_f16(false, af, false, bf1, (short)0, acc1,
                                                  false, false);
  }
  if (Kmain < C) {
    v16h af  = load_a_tail(Arow,  N, Kmain, C, half);
    v16h bf0 = load_b_tail(Bcol0, N, Kmain, C, half);
    v16h bf1 = load_b_tail(Bcol1, N, Kmain, C, half);
    acc0 = __builtin_amdgcn_wmma_f32_16x16x32_f16(false, af, false, bf0, (short)0, acc0,
                                                  false, false);
    acc1 = __builtin_amdgcn_wmma_f32_16x16x32_f16(false, af, false, bf1, (short)0, acc1,
                                                  false, false);
  }
  float xn0 = xx[n0], xn1 = xx[n1];
#pragma unroll
  for (int r = 0; r < 8; ++r) {
    int mo = tm * 16 + r + half * 8;
    float xm = xx[mo];
    P[(long)mo * N + n0] = 2.0f * acc0[r] - xm - xn0;
    P[(long)mo * N + n1] = 2.0f * acc1[r] - xm - xn1;
  }
}

// ---------------- squared norms per point -----------------------------
__global__ void gda_sqnorm_kernel(const float* __restrict__ X, float* __restrict__ xx,
                                  int C, int N, int total) {
  int g = blockIdx.x * 256 + threadIdx.x;
  if (g >= total) return;
  int b = g / N, n = g % N;
  const float* Xb = X + (long)b * C * N;
  float s = 0.f;
  for (int c = 0; c < C; ++c) { float v = Xb[(long)c * N + n]; s += v * v; }
  xx[g] = s;
}

// ---------------- iterative top-k (descending, lowest index on ties) -------------------
__global__ void gda_topk_kernel(const float* __restrict__ vals, long rowStride,
                                int len, int k, int* __restrict__ outIdx, int negate)
{
  __shared__ float sv[1024];
  __shared__ float rv[256];
  __shared__ int   ri[256];
  int row = blockIdx.x;
  int tid = threadIdx.x;
  const float* v = vals + (long)row * rowStride;
  for (int i = tid; i < len; i += 256) sv[i] = negate ? -v[i] : v[i];
  __syncthreads();
  for (int it = 0; it < k; ++it) {
    float best = -FLT_MAX; int bi = len;
    for (int i = tid; i < len; i += 256) {
      float x = sv[i];
      if (x > best) { best = x; bi = i; }
    }
    rv[tid] = best; ri[tid] = bi;
    __syncthreads();
    for (int off = 128; off > 0; off >>= 1) {
      if (tid < off) {
        float ov = rv[tid + off]; int oi = ri[tid + off];
        if (ov > rv[tid] || (ov == rv[tid] && oi < ri[tid])) { rv[tid] = ov; ri[tid] = oi; }
      }
      __syncthreads();
    }
    if (tid == 0) { outIdx[(long)row * k + it] = ri[0]; sv[ri[0]] = -FLT_MAX; }
    __syncthreads();
  }
}

// ---------------- GDM: column sums of w = mask * exp(p/4) ------------------------------
__global__ void gda_colsum_kernel(const float* __restrict__ p, float* __restrict__ s, int N) {
  int m = blockIdx.x * 256 + threadIdx.x;
  if (m >= N) return;
  float acc = 0.f;
  for (int n = 0; n < N; ++n) {
    float v = p[(long)n * N + m];
    if (fabsf(v) < 0.04f) acc += expf(v * 0.25f);   // sqrt(|p|)<0.2, exp(p/sigma^2)
  }
  s[m] = acc;
}

// ---------------- GDM: pai[n] = ||x_n - sum_j A[n,idx_j] * x_idx_j||^2  (C==64) --------
__global__ void gda_pai_kernel(const float* __restrict__ p, const float* __restrict__ s,
                               const int* __restrict__ idx64, const float* __restrict__ xb,
                               float* __restrict__ pai, int N)
{
  int n = blockIdx.x;
  int t = threadIdx.x;                 // 64 threads = channels
  __shared__ float aj[63];
  __shared__ int   mj[63];
  __shared__ float red[64];
  if (t < 63) {
    int m = idx64[(long)n * 64 + t + 1];
    mj[t] = m;
    float v = p[(long)n * N + m];
    float w = (fabsf(v) < 0.04f) ? expf(v * 0.25f) : 0.f;
    aj[t] = w / s[n];
  }
  __syncthreads();
  float nv = 0.f;
  for (int j = 0; j < 63; ++j) nv += aj[j] * xb[(long)t * N + mj[j]];
  float d = xb[(long)t * N + n] - nv;
  red[t] = d * d;
  __syncthreads();
  if (t == 0) {
    float acc = 0.f;
    for (int i = 0; i < 64; ++i) acc += red[i];
    pai[n] = acc;
  }
}

// ---------------- gather selected points: out[b,c,i] = X[b,c,sel[b,i]] ------------------
__global__ void gda_gather_kernel(const float* __restrict__ X, const int* __restrict__ sel,
                                  float* __restrict__ out, int C, int N, int M, int total) {
  int g = blockIdx.x * 256 + threadIdx.x;
  if (g >= total) return;
  int i = g % M;
  int c = (g / M) % C;
  int b = g / (M * C);
  out[((long)b * C + c) * M + i] = X[((long)b * C + c) * N + sel[(long)b * M + i]];
}

// ---------------- fold edge weights: G = W1+W2, keep W1 ---------------------------------
__global__ void gda_splitw_kernel(const float* __restrict__ W, float* __restrict__ G,
                                  float* __restrict__ W1, int O, int C) {
  int g = blockIdx.x * 256 + threadIdx.x;
  if (g >= O * C) return;
  int o = g / C, c = g % C;
  float w1 = W[(long)o * 2 * C + c];
  float w2 = W[(long)o * 2 * C + C + c];
  G[g] = w1 + w2;
  W1[g] = w1;
}

// edge conv epilogue: relu1[o, n*30+k] = relu(bns[o]*(y[o,idx]-t[o,n]+cb[o])+bnb[o])
__global__ void gda_edge_kernel(const float* __restrict__ y, const float* __restrict__ t,
                                const float* __restrict__ cb, const float* __restrict__ bns,
                                const float* __restrict__ bnb, const int* __restrict__ idx30,
                                float* __restrict__ out, int O, int N) {
  int g = blockIdx.x * 256 + threadIdx.x;
  int total = O * N * 30;
  if (g >= total) return;
  int kk = g % 30;
  int n = (g / 30) % N;
  int o = g / (30 * N);
  int m = idx30[(long)n * 30 + kk];
  float v = y[(long)o * N + m] - t[(long)o * N + n] + cb[o];
  v = bns[o] * v + bnb[o];
  out[(long)o * (N * 30) + (long)n * 30 + kk] = fmaxf(v, 0.f);
}

__global__ void gda_maxk_kernel(const float* __restrict__ v2, float* __restrict__ out,
                                int O, int N) {
  int g = blockIdx.x * 256 + threadIdx.x;
  if (g >= O * N) return;
  int n = g % N, o = g / N;
  float m = -FLT_MAX;
  for (int k = 0; k < 30; ++k) m = fmaxf(m, v2[(long)o * N * 30 + (long)n * 30 + k]);
  out[(long)o * N + n] = m;
}

__global__ void gda_copy_kernel(float* __restrict__ dst, const float* __restrict__ src,
                                long count, long dstB, long srcB, long total) {
  long g = (long)blockIdx.x * 256 + threadIdx.x;
  if (g >= total) return;
  long b = g / count, i = g % count;
  dst[b * dstB + i] = src[b * srcB + i];
}

__global__ void gda_pool_kernel(const float* __restrict__ h4, float* __restrict__ out,
                                int B, int C, int N) {
  int g = blockIdx.x * 256 + threadIdx.x;
  if (g >= B * C) return;
  int o = g % C, b = g / C;
  const float* r = h4 + ((long)b * C + o) * N;
  float mx = -FLT_MAX, sm = 0.f;
  for (int n = 0; n < N; ++n) { float v = r[n]; mx = fmaxf(mx, v); sm += v; }
  out[(long)b * 2 * C + o] = mx;
  out[(long)b * 2 * C + C + o] = sm * (1.0f / N);
}

// ---------------- host-side GEMM launcher ----------------------------------------------
static void launch_gemm(hipStream_t st, int batches,
                        const float* A, int sam, int sak, long sab,
                        const float* Bp, int sbk, int sbn, long sbb,
                        float* C, int scm, int scn, long scb,
                        const float* bias, const float* bns, const float* bnb,
                        const float* R, int srm, int srn, long srb,
                        int M, int N, int K, float alpha, int relu) {
  int tilesN2 = (((N + 15) / 16) + 1) / 2;
  int tiles = ((M + 15) / 16) * tilesN2;
  dim3 g((tiles + 3) / 4, batches, 1), b(32, 4, 1);
  gda_wmma_gemm_kernel<<<g, b, 0, st>>>(A, sam, sak, sab, Bp, sbk, sbn, sbb,
                                        C, scm, scn, scb, bias, bns, bnb,
                                        R, srm, srn, srb, M, N, K, alpha, relu);
}

extern "C" void kernel_launch(void* const* d_in, const int* in_sizes, int n_in,
                              void* d_out, int out_size, void* d_ws, size_t ws_size,
                              hipStream_t stream) {
  (void)in_sizes; (void)n_in; (void)out_size; (void)ws_size;
  const int B = NB, N = NPTS;
  auto F = [&](int i) { return (const float*)d_in[i]; };
  const float* x = F(0);   // (16,3,1024)
  // param indices:
  // 1 conv1_w 2 conv1_b 3 bn1_s 4 bn1_b | 5 conv11 .. 8 | 9 conv12 .. 12 | 13 conv2 .. 16
  // 17 conv21 .. 20 | 21 conv22 .. 24 | 25 conv3 .. 28 | 29 conv31 .. 32 | 33 conv32 .. 36
  // 37 conv4 .. 40 | 41 sg1s(10) 51 sg1g 61 sg2s 71 sg2g | 81 l1_w 82 l1_b 83 l2_w 84 l2_b 85 l3_w 86 l3_b

  char* base = (char*)d_ws;
  size_t off = 0;
  auto allocF = [&](size_t e) { float* p = (float*)(base + off); off += ((e * 4 + 255) / 256) * 256; return p; };
  auto allocI = [&](size_t e) { int* p = (int*)(base + off); off += ((e * 4 + 255) / 256) * 256; return p; };

  float* p1    = allocF((size_t)N * N);           // per-batch pairwise p
  int*   idx30 = allocI((size_t)B * N * 30);
  int*   idx64 = allocI((size_t)B * N * 64);
  float* xxb   = allocF((size_t)B * N);
  float* scol  = allocF((size_t)B * N);
  float* pai   = allocF((size_t)B * N);
  int*   pais  = allocI((size_t)B * MGDM);
  int*   paig  = allocI((size_t)B * MGDM);
  float* Gw    = allocF(128 * 131);
  float* W1w   = allocF(128 * 131);
  float* yE    = allocF((size_t)B * 128 * N);
  float* tE    = allocF((size_t)B * 128 * N);
  float* relu1 = allocF((size_t)128 * N * 30);    // per-batch
  float* v2    = allocF((size_t)128 * N * 30);    // per-batch
  float* x1    = allocF((size_t)B * 64 * N);
  float* x2f   = allocF((size_t)B * 64 * N);
  float* x3f   = allocF((size_t)B * 128 * N);
  float* z1    = allocF((size_t)B * 64 * N);
  float* z2    = allocF((size_t)B * 64 * N);
  float* z3    = allocF((size_t)B * 128 * N);
  float* x1t   = allocF((size_t)B * 67 * N);
  float* x2t   = allocF((size_t)B * 131 * N);
  float* xs    = allocF((size_t)B * 64 * MGDM);
  float* xg    = allocF((size_t)B * 64 * MGDM);
  float* gg    = allocF((size_t)B * 32 * MGDM);
  float* tt    = allocF((size_t)B * 32 * N);
  float* phb   = allocF((size_t)B * 32 * MGDM);
  float* Watt  = allocF((size_t)B * N * MGDM);
  float* yT    = allocF((size_t)B * 32 * N);
  float* ys    = allocF((size_t)B * 64 * N);
  float* yg    = allocF((size_t)B * 64 * N);
  float* caty  = allocF((size_t)B * 128 * N);
  float* hbuf  = allocF((size_t)B * 256 * N);
  float* h4    = allocF((size_t)B * 512 * N);
  float* hpool = allocF((size_t)B * 1024);
  float* fc1   = allocF((size_t)512 * 16);
  float* fc2   = allocF((size_t)256 * 16);

  // ---- local_operator + conv + conv1x1 + max(k) stage ----
  auto edge_stage = [&](const float* feat, int C, int wA, int O1, int wB, int O2, float* outX) {
    int total = B * N;
    gda_sqnorm_kernel<<<dim3((total + 255) / 256), 256, 0, stream>>>(feat, xxb, C, N, total);
    int tiles = (N / 16) * (N / 32);
    for (int b = 0; b < B; ++b) {
      gda_pmat_kernel<<<dim3((tiles + 3) / 4), dim3(32, 4), 0, stream>>>(
          feat + (size_t)b * C * N, xxb + (size_t)b * N, p1, C, N);
      gda_topk_kernel<<<dim3(N), 256, 0, stream>>>(p1, N, N, 30, idx30 + (size_t)b * N * 30, 0);
    }
    gda_splitw_kernel<<<dim3((O1 * C + 255) / 256), 256, 0, stream>>>(F(wA), Gw, W1w, O1, C);
    launch_gemm(stream, B, Gw, C, 1, 0, feat, N, 1, (long)C * N, yE, N, 1, (long)O1 * N,
                nullptr, nullptr, nullptr, nullptr, 0, 0, 0, O1, N, C, 1.f, 0);
    launch_gemm(stream, B, W1w, C, 1, 0, feat, N, 1, (long)C * N, tE, N, 1, (long)O1 * N,
                nullptr, nullptr, nullptr, nullptr, 0, 0, 0, O1, N, C, 1.f, 0);
    for (int b = 0; b < B; ++b) {
      int tot = O1 * N * 30;
      gda_edge_kernel<<<dim3((tot + 255) / 256), 256, 0, stream>>>(
          yE + (size_t)b * O1 * N, tE + (size_t)b * O1 * N, F(wA + 1), F(wA + 2), F(wA + 3),
          idx30 + (size_t)b * N * 30, relu1, O1, N);
      launch_gemm(stream, 1, F(wB), O1, 1, 0, relu1, N * 30, 1, 0,
                  v2, N * 30, 1, 0, F(wB + 1), F(wB + 2), F(wB + 3),
                  nullptr, 0, 0, 0, O2, N * 30, O1, 1.f, 1);
      gda_maxk_kernel<<<dim3((O2 * N + 255) / 256), 256, 0, stream>>>(
          v2, outX + (size_t)b * O2 * N, O2, N);
    }
  };

  // ---- SGCAM: out = bn(w_w * (gg * W^T)) + xf,  W = (tt^T * ph)/M ----
  auto sgcam = [&](const float* xf, const float* x2, int pb, float* out) {
    launch_gemm(stream, B, F(pb), 64, 1, 0, x2, MGDM, 1, (long)64 * MGDM,
                gg, MGDM, 1, (long)32 * MGDM, F(pb + 1), nullptr, nullptr,
                nullptr, 0, 0, 0, 32, MGDM, 64, 1.f, 0);
    launch_gemm(stream, B, F(pb + 2), 64, 1, 0, xf, N, 1, (long)64 * N,
                tt, N, 1, (long)32 * N, F(pb + 3), nullptr, nullptr,
                nullptr, 0, 0, 0, 32, N, 64, 1.f, 0);
    launch_gemm(stream, B, F(pb + 4), 64, 1, 0, x2, MGDM, 1, (long)64 * MGDM,
                phb, MGDM, 1, (long)32 * MGDM, F(pb + 5), nullptr, nullptr,
                nullptr, 0, 0, 0, 32, MGDM, 64, 1.f, 0);
    launch_gemm(stream, B, tt, 1, N, (long)32 * N, phb, MGDM, 1, (long)32 * MGDM,
                Watt, MGDM, 1, (long)N * MGDM, nullptr, nullptr, nullptr,
                nullptr, 0, 0, 0, N, MGDM, 32, 1.0f / MGDM, 0);
    launch_gemm(stream, B, gg, MGDM, 1, (long)32 * MGDM, Watt, 1, MGDM, (long)N * MGDM,
                yT, N, 1, (long)32 * N, nullptr, nullptr, nullptr,
                nullptr, 0, 0, 0, 32, N, MGDM, 1.f, 0);
    launch_gemm(stream, B, F(pb + 6), 32, 1, 0, yT, N, 1, (long)32 * N,
                out, N, 1, (long)64 * N, F(pb + 7), F(pb + 8), F(pb + 9),
                xf, N, 1, (long)64 * N, 64, N, 32, 1.f, 0);
  };

  // ---- GDM + 2x SGCAM + concat-conv stage ----
  auto gdm_stage = [&](const float* xf, int sgS, int sgG, int wCat, float* zout) {
    int total = B * N;
    gda_sqnorm_kernel<<<dim3((total + 255) / 256), 256, 0, stream>>>(xf, xxb, 64, N, total);
    int tiles = (N / 16) * (N / 32);
    for (int b = 0; b < B; ++b) {
      const float* xb = xf + (size_t)b * 64 * N;
      gda_pmat_kernel<<<dim3((tiles + 3) / 4), dim3(32, 4), 0, stream>>>(
          xb, xxb + (size_t)b * N, p1, 64, N);
      gda_topk_kernel<<<dim3(N), 256, 0, stream>>>(p1, N, N, 64, idx64 + (size_t)b * N * 64, 0);
      gda_colsum_kernel<<<dim3((N + 255) / 256), 256, 0, stream>>>(p1, scol + (size_t)b * N, N);
      gda_pai_kernel<<<dim3(N), 64, 0, stream>>>(p1, scol + (size_t)b * N,
                                                 idx64 + (size_t)b * N * 64, xb,
                                                 pai + (size_t)b * N, N);
    }
    gda_topk_kernel<<<dim3(B), 256, 0, stream>>>(pai, N, N, MGDM, pais, 0);
    gda_topk_kernel<<<dim3(B), 256, 0, stream>>>(pai, N, N, MGDM, paig, 1);
    int tg = B * 64 * MGDM;
    gda_gather_kernel<<<dim3((tg + 255) / 256), 256, 0, stream>>>(xf, pais, xs, 64, N, MGDM, tg);
    gda_gather_kernel<<<dim3((tg + 255) / 256), 256, 0, stream>>>(xf, paig, xg, 64, N, MGDM, tg);
    sgcam(xf, xs, sgS, ys);
    sgcam(xf, xg, sgG, yg);
    long tc = (long)B * 64 * N;
    gda_copy_kernel<<<dim3((tc + 255) / 256), 256, 0, stream>>>(caty, ys, 64 * N, 128 * N, 64 * N, tc);
    gda_copy_kernel<<<dim3((tc + 255) / 256), 256, 0, stream>>>(caty + (size_t)64 * N, yg, 64 * N, 128 * N, 64 * N, tc);
    launch_gemm(stream, B, F(wCat), 128, 1, 0, caty, N, 1, (long)128 * N,
                zout, N, 1, (long)64 * N, F(wCat + 1), F(wCat + 2), F(wCat + 3),
                nullptr, 0, 0, 0, 64, N, 128, 1.f, 1);
  };

  long tc;
  // ======== Stage 1 ========
  edge_stage(x, 3, 1, 64, 5, 64, x1);
  gdm_stage(x1, 41, 51, 9, z1);
  tc = (long)B * 3 * N;
  gda_copy_kernel<<<dim3((tc + 255) / 256), 256, 0, stream>>>(x1t, x, 3 * N, 67 * N, 3 * N, tc);
  tc = (long)B * 64 * N;
  gda_copy_kernel<<<dim3((tc + 255) / 256), 256, 0, stream>>>(x1t + (size_t)3 * N, z1, 64 * N, 67 * N, 64 * N, tc);

  // ======== Stage 2 ========
  edge_stage(x1t, 67, 13, 64, 17, 64, x2f);
  gdm_stage(x2f, 61, 71, 21, z2);
  tc = (long)B * 67 * N;
  gda_copy_kernel<<<dim3((tc + 255) / 256), 256, 0, stream>>>(x2t, x1t, 67 * N, 131 * N, 67 * N, tc);
  tc = (long)B * 64 * N;
  gda_copy_kernel<<<dim3((tc + 255) / 256), 256, 0, stream>>>(x2t + (size_t)67 * N, z2, 64 * N, 131 * N, 64 * N, tc);

  // ======== Stage 3 ========
  edge_stage(x2t, 131, 25, 128, 29, 128, x3f);
  launch_gemm(stream, B, F(33), 128, 1, 0, x3f, N, 1, (long)128 * N,
              z3, N, 1, (long)128 * N, F(34), F(35), F(36),
              nullptr, 0, 0, 0, 128, N, 128, 1.f, 1);

  // ======== Head ========
  tc = (long)B * 64 * N;
  gda_copy_kernel<<<dim3((tc + 255) / 256), 256, 0, stream>>>(hbuf, z1, 64 * N, 256 * N, 64 * N, tc);
  gda_copy_kernel<<<dim3((tc + 255) / 256), 256, 0, stream>>>(hbuf + (size_t)64 * N, z2, 64 * N, 256 * N, 64 * N, tc);
  tc = (long)B * 128 * N;
  gda_copy_kernel<<<dim3((tc + 255) / 256), 256, 0, stream>>>(hbuf + (size_t)128 * N, z3, 128 * N, 256 * N, 128 * N, tc);

  launch_gemm(stream, B, F(37), 256, 1, 0, hbuf, N, 1, (long)256 * N,
              h4, N, 1, (long)512 * N, F(38), F(39), F(40),
              nullptr, 0, 0, 0, 512, N, 256, 1.f, 1);
  gda_pool_kernel<<<dim3((B * 512 + 255) / 256), 256, 0, stream>>>(h4, hpool, B, 512, N);

  // FC head: treat batch as GEMM N dimension
  launch_gemm(stream, 1, F(81), 1024, 1, 0, hpool, 1, 1024, 0, fc1, 16, 1, 0,
              F(82), nullptr, nullptr, nullptr, 0, 0, 0, 512, 16, 1024, 1.f, 1);
  launch_gemm(stream, 1, F(83), 512, 1, 0, fc1, 16, 1, 0, fc2, 16, 1, 0,
              F(84), nullptr, nullptr, nullptr, 0, 0, 0, 256, 16, 512, 1.f, 1);
  launch_gemm(stream, 1, F(85), 256, 1, 0, fc2, 16, 1, 0, (float*)d_out, 1, 20, 0,
              F(86), nullptr, nullptr, nullptr, 0, 0, 0, 20, 16, 256, 1.f, 0);
}
